// MultiQueryAttention_48799418417749
// MI455X (gfx1250) — matrix operand order repaired and checked by reading, not compiled, not run on hardware
//
#include <hip/hip_runtime.h>
#include <hip/hip_bf16.h>
#include <stdint.h>

#define N_HEADS 16
#define BDIM 2
#define TDIM 2048
#define CDIM 1024
#define DDIM 64

typedef __bf16 bf16_t;
typedef __bf16 v16bf __attribute__((ext_vector_type(16)));
typedef float  v8f   __attribute__((ext_vector_type(8)));
typedef uint32_t u32x4 __attribute__((ext_vector_type(4)));

union FragAB { v16bf v; uint32_t u[8]; bf16_t h[16]; };
union FragC  { v8f  v; float f[8]; };

// ---------------------------------------------------------------------------
// Async global -> LDS copy (CDNA5 GLOBAL_LOAD_ASYNC_TO_LDS_B128, ASYNCcnt).
// Builtin takes int4-vector pointers in AS1 (global) / AS3 (LDS).
// Guarded by __has_builtin with a synchronous fallback so compiles either way.
// ---------------------------------------------------------------------------
#if defined(__has_builtin)
#if __has_builtin(__builtin_amdgcn_global_load_async_to_lds_b128) && \
    __has_builtin(__builtin_amdgcn_s_wait_asynccnt)
#define HAVE_ASYNC_LDS 1
#endif
#endif

#if defined(HAVE_ASYNC_LDS)
typedef int v4i_t __attribute__((vector_size(16)));
typedef __attribute__((address_space(1))) v4i_t as1_v4i;
typedef __attribute__((address_space(3))) v4i_t as3_v4i;
#endif

static __device__ __forceinline__ void async_copy_b128(const void* g, void* l) {
#if defined(HAVE_ASYNC_LDS)
    __builtin_amdgcn_global_load_async_to_lds_b128(
        (as1_v4i*)(uintptr_t)g, (as3_v4i*)(uint32_t)(uintptr_t)l, 0, 0);
#else
    *(u32x4*)l = *(const u32x4*)g;
#endif
}

static __device__ __forceinline__ void async_wait_all() {
#if defined(HAVE_ASYNC_LDS)
    __builtin_amdgcn_s_wait_asynccnt(0);
#endif
}

// ---------------------------------------------------------------------------
// WMMA fragment loaders (wave32; A and B both use contiguous k = lh*16 + s order)
// ---------------------------------------------------------------------------

// A-matrix 16x32 bf16: lane l holds row m=l&15; lane-half lh=l>>4;
// dword j holds K = lh*16 + j*2 (+0,+1).  base row-major, ld elems.
static __device__ __forceinline__ v16bf load_a_frag(const bf16_t* base, int ld) {
    const int lane = threadIdx.x & 31;
    const int m  = lane & 15;
    const int lh = lane >> 4;
    FragAB fa;
#pragma unroll
    for (int j = 0; j < 8; ++j) {
        int k = (lh << 4) | (j << 1);
        fa.u[j] = *(const uint32_t*)(base + m * ld + k);
    }
    return fa.v;
}

// B-matrix 32x16 bf16, element (k,n) at base[n*ld + k] (K contiguous per column).
static __device__ __forceinline__ v16bf load_b_kcontig(const bf16_t* base, int ld) {
    const int lane = threadIdx.x & 31;
    const int n  = lane & 15;
    const int lh = lane >> 4;
    FragAB fb;
#pragma unroll
    for (int j = 0; j < 8; ++j) {
        int k = (lh << 4) | (j << 1);
        fb.u[j] = *(const uint32_t*)(base + n * ld + k);
    }
    return fb.v;
}

// B-matrix 32x16 bf16, element (k,n) at base[k*ld + n] (row-major K x N source).
static __device__ __forceinline__ v16bf load_b_ncontig(const bf16_t* base, int ld) {
    const int lane = threadIdx.x & 31;
    const int n  = lane & 15;
    const int lh = lane >> 4;
    FragAB fb;
#pragma unroll
    for (int j = 0; j < 8; ++j) {
        int k0 = (lh << 4) | (j << 1);
        fb.h[2 * j]     = base[k0 * ld + n];
        fb.h[2 * j + 1] = base[(k0 + 1) * ld + n];
    }
    return fb.v;
}

// ---------------------------------------------------------------------------
// f32 -> bf16 conversion
// ---------------------------------------------------------------------------
__global__ void cvt_f32_bf16_k(const float* __restrict__ in, bf16_t* __restrict__ out, int n) {
    int i = blockIdx.x * 256 + threadIdx.x;
    if (i < n) out[i] = (bf16_t)in[i];
}

// ---------------------------------------------------------------------------
// Tiled bf16 GEMM: C[M,N] = A[M,K] * B[K,N].  BM=BN=64, BK=32, 128 threads.
// A/B tiles staged via async LDS DMA; wave w computes rows w*16..w*16+15.
// ---------------------------------------------------------------------------
template <bool F32OUT>
__global__ __launch_bounds__(128) void gemm_bf16_k(
    const bf16_t* __restrict__ A, const bf16_t* __restrict__ Bw,
    bf16_t* __restrict__ Cb, float* __restrict__ Cf, int M, int N, int K) {
    __shared__ alignas(16) bf16_t sA[64][32];   // row-major M x K
    __shared__ alignas(16) bf16_t sB[32][64];   // row-major K x N

    const int tid  = threadIdx.x;
    const int wid  = tid >> 5;
    const int lane = tid & 31;
    const int mBase = blockIdx.y * 64;
    const int nBase = blockIdx.x * 64;

    FragC acc[4];
#pragma unroll
    for (int i = 0; i < 4; ++i) acc[i].v = (v8f){0, 0, 0, 0, 0, 0, 0, 0};

    for (int kt = 0; kt < K; kt += 32) {
        // A tile: 64x32 bf16 = 256 x 16B chunks (4 chunks per 32-elem row)
#pragma unroll
        for (int i = 0; i < 2; ++i) {
            int ch = tid + i * 128;
            int r = ch >> 2, e = (ch & 3) << 3;
            async_copy_b128(A + (size_t)(mBase + r) * K + kt + e, &sA[r][e]);
        }
        // B tile: 32x64 bf16 = 256 x 16B chunks (8 chunks per 64-elem row)
#pragma unroll
        for (int i = 0; i < 2; ++i) {
            int ch = tid + i * 128;
            int r = ch >> 3, e = (ch & 7) << 3;
            async_copy_b128(Bw + (size_t)(kt + r) * N + nBase + e, &sB[r][e]);
        }
        if (kt + 32 < K) {  // speculative prefetch of next A tile
            __builtin_prefetch(A + (size_t)(mBase + (tid >> 1)) * K + kt + 32, 0, 1);
        }
        async_wait_all();
        __syncthreads();

        v16bf a = load_a_frag(&sA[wid * 16][0], 32);
#pragma unroll
        for (int nt = 0; nt < 4; ++nt) {
            v16bf b = load_b_ncontig(&sB[0][nt * 16], 64);
            acc[nt].v = __builtin_amdgcn_wmma_f32_16x16x32_bf16(
                false, a, false, b, (short)0, acc[nt].v, false, false);
        }
        __syncthreads();
    }

    const int lh = lane >> 4, nn = lane & 15;
#pragma unroll
    for (int nt = 0; nt < 4; ++nt) {
#pragma unroll
        for (int j = 0; j < 8; ++j) {
            size_t r = (size_t)(mBase + wid * 16 + lh * 8 + j);
            size_t c = (size_t)(nBase + nt * 16 + nn);
            if (F32OUT) Cf[r * N + c] = acc[nt].f[j];
            else        Cb[r * N + c] = (bf16_t)acc[nt].f[j];
        }
    }
}

// ---------------------------------------------------------------------------
// RoPE.  q: [B,T,H*D] bf16 -> [B,H,T,D] bf16, scaled by D^-0.5.
// pair (d, d+32) within each 64-wide head; inv_freq = 10000^(-d/32).
// ---------------------------------------------------------------------------
__global__ void rope_q_k(const bf16_t* __restrict__ qin, bf16_t* __restrict__ qout) {
    int idx = blockIdx.x * 256 + threadIdx.x;          // B*T*H*32 threads
    int d = idx & 31;
    int h = (idx >> 5) & 15;
    int t = (idx >> 9) & (TDIM - 1);
    int b = idx >> 20;
    float inv = __expf(-(float)d * (9.2103403719761836f / 32.0f));
    float ang = (float)t * inv;
    float s, c;
    __sincosf(ang, &s, &c);
    size_t inRow = ((size_t)(b * TDIM + t)) * CDIM + (size_t)h * DDIM;
    float q0 = (float)qin[inRow + d];
    float q1 = (float)qin[inRow + d + 32];
    const float scale = 0.125f;  // 64^-0.5
    size_t outRow = (((size_t)b * N_HEADS + h) * TDIM + t) * DDIM;
    qout[outRow + d]      = (bf16_t)((q0 * c - q1 * s) * scale);
    qout[outRow + d + 32] = (bf16_t)((q1 * c + q0 * s) * scale);
}

__global__ void rope_k_k(const bf16_t* __restrict__ kin, bf16_t* __restrict__ kout) {
    int idx = blockIdx.x * 256 + threadIdx.x;          // B*T*32 threads
    int d = idx & 31;
    int t = (idx >> 5) & (TDIM - 1);
    int b = idx >> 16;
    float inv = __expf(-(float)d * (9.2103403719761836f / 32.0f));
    float ang = (float)t * inv;
    float s, c;
    __sincosf(ang, &s, &c);
    size_t row = ((size_t)(b * TDIM + t)) * DDIM;
    float k0 = (float)kin[row + d];
    float k1 = (float)kin[row + d + 32];
    kout[row + d]      = (bf16_t)(k0 * c - k1 * s);
    kout[row + d + 32] = (bf16_t)(k1 * c + k0 * s);
}

// ---------------------------------------------------------------------------
// Flash-attention MQA, causal.  One block = (b, h, 64 q-rows); 4 waves,
// wave w owns 16 q-rows.  K/V streamed in 32-key LDS tiles via async DMA.
// S = Q K^T via wmma (bf16 in, f32 acc), online softmax in f32,
// P -> LDS -> A-fragment, O += P V via wmma.
// ---------------------------------------------------------------------------
__global__ __launch_bounds__(128) void fa_mqa_k(
    const bf16_t* __restrict__ qr, const bf16_t* __restrict__ kr,
    const bf16_t* __restrict__ vb, bf16_t* __restrict__ attn) {
    __shared__ alignas(16) bf16_t sK[32][64];
    __shared__ alignas(16) bf16_t sV[32][64];
    __shared__ alignas(16) bf16_t sP[4][16][32];

    const int tid  = threadIdx.x;
    const int wid  = tid >> 5;
    const int lane = tid & 31;
    const int lh = lane >> 4, nn = lane & 15;
    const int qb = blockIdx.x, hh = blockIdx.y, bb = blockIdx.z;
    const int q0 = qb * 64;

    // Q fragments (already RoPE'd and pre-scaled by D^-0.5), from [B,H,T,D]
    const bf16_t* qbase =
        qr + (((size_t)bb * N_HEADS + hh) * TDIM + q0 + wid * 16) * DDIM;
    v16bf qa0 = load_a_frag(qbase, DDIM);       // d = 0..31
    v16bf qa1 = load_a_frag(qbase + 32, DDIM);  // d = 32..63

    FragC accO[4];
#pragma unroll
    for (int i = 0; i < 4; ++i) accO[i].v = (v8f){0, 0, 0, 0, 0, 0, 0, 0};
    float mI[8], lI[8];
#pragma unroll
    for (int j = 0; j < 8; ++j) { mI[j] = -1e30f; lI[j] = 0.0f; }

    const bf16_t* kbaseG = kr + (size_t)bb * TDIM * DDIM;
    const bf16_t* vbaseG = vb + (size_t)bb * TDIM * DDIM;

    const int ntiles = (q0 + 64) / 32;  // causal: only tiles touching <= last q row
    for (int kt = 0; kt < ntiles; ++kt) {
        const int kbase = kt * 32;
        // K and V tiles: 32x64 bf16 each = 256 x 16B chunks each
#pragma unroll
        for (int i = 0; i < 2; ++i) {
            int ch = tid + i * 128;
            int r = ch >> 3, e = (ch & 7) << 3;
            size_t goff = (size_t)(kbase + r) * DDIM + e;
            async_copy_b128(kbaseG + goff, &sK[r][e]);
            async_copy_b128(vbaseG + goff, &sV[r][e]);
        }
        async_wait_all();
        __syncthreads();

        // S tile: 16 q-rows x 32 keys per wave -> two 16x16 C fragments
        FragC sacc[2];
        sacc[0].v = (v8f){0, 0, 0, 0, 0, 0, 0, 0};
        sacc[1].v = (v8f){0, 0, 0, 0, 0, 0, 0, 0};
#pragma unroll
        for (int nt = 0; nt < 2; ++nt) {
            v16bf bk0 = load_b_kcontig(&sK[nt * 16][0], 64);   // d 0..31
            sacc[nt].v = __builtin_amdgcn_wmma_f32_16x16x32_bf16(
                false, qa0, false, bk0, (short)0, sacc[nt].v, false, false);
            v16bf bk1 = load_b_kcontig(&sK[nt * 16][32], 64);  // d 32..63
            sacc[nt].v = __builtin_amdgcn_wmma_f32_16x16x32_bf16(
                false, qa1, false, bk1, (short)0, sacc[nt].v, false, false);
        }

        // causal mask + online softmax (rows m = lh*8+j, key col = nt*16+nn)
        const int qrow = q0 + wid * 16 + lh * 8;
#pragma unroll
        for (int j = 0; j < 8; ++j) {
            int qp = qrow + j;
            float s0 = (kbase + nn      <= qp) ? sacc[0].f[j] : -1e30f;
            float s1 = (kbase + 16 + nn <= qp) ? sacc[1].f[j] : -1e30f;
            float rmax = fmaxf(s0, s1);
            rmax = fmaxf(rmax, __shfl_xor(rmax, 1, 32));
            rmax = fmaxf(rmax, __shfl_xor(rmax, 2, 32));
            rmax = fmaxf(rmax, __shfl_xor(rmax, 4, 32));
            rmax = fmaxf(rmax, __shfl_xor(rmax, 8, 32));
            float mn   = fmaxf(mI[j], rmax);
            float corr = __expf(mI[j] - mn);
            mI[j] = mn;
            float p0 = __expf(s0 - mn);
            float p1 = __expf(s1 - mn);
            float rs = p0 + p1;
            rs += __shfl_xor(rs, 1, 32);
            rs += __shfl_xor(rs, 2, 32);
            rs += __shfl_xor(rs, 4, 32);
            rs += __shfl_xor(rs, 8, 32);
            lI[j] = lI[j] * corr + rs;
            accO[0].f[j] *= corr;
            accO[1].f[j] *= corr;
            accO[2].f[j] *= corr;
            accO[3].f[j] *= corr;
            sP[wid][lh * 8 + j][nn]      = (bf16_t)p0;
            sP[wid][lh * 8 + j][16 + nn] = (bf16_t)p1;
        }

        // O += P * V   (per-wave LDS round-trip converts C-layout -> A-layout)
        v16bf pa = load_a_frag(&sP[wid][0][0], 32);
#pragma unroll
        for (int nt = 0; nt < 4; ++nt) {
            v16bf bv = load_b_ncontig(&sV[0][nt * 16], 64);
            accO[nt].v = __builtin_amdgcn_wmma_f32_16x16x32_bf16(
                false, pa, false, bv, (short)0, accO[nt].v, false, false);
        }
        __syncthreads();
    }

    // epilogue: O /= l, store into attn [B*T, H*D] (head-transposed back)
    const size_t orow = (size_t)bb * TDIM + q0 + wid * 16 + lh * 8;
#pragma unroll
    for (int j = 0; j < 8; ++j) {
        float invl = 1.0f / lI[j];
        size_t rbase = (orow + j) * CDIM + (size_t)hh * DDIM;
#pragma unroll
        for (int nt = 0; nt < 4; ++nt)
            attn[rbase + nt * 16 + nn] = (bf16_t)(accO[nt].f[j] * invl);
    }
}

// ---------------------------------------------------------------------------
// Host-side orchestration
// ---------------------------------------------------------------------------
extern "C" void kernel_launch(void* const* d_in, const int* in_sizes, int n_in,
                              void* d_out, int out_size, void* d_ws, size_t ws_size,
                              hipStream_t stream) {
    (void)in_sizes; (void)n_in; (void)out_size; (void)ws_size;
    const float* x  = (const float*)d_in[0];
    const float* Wq = (const float*)d_in[1];
    const float* Wk = (const float*)d_in[2];
    const float* Wv = (const float*)d_in[3];
    const float* Wo = (const float*)d_in[4];
    float* out = (float*)d_out;

    const int M = BDIM * TDIM;  // 4096
    char* ws = (char*)d_ws;
    size_t off = 0;
    auto carve = [&](size_t bytes) -> char* {
        char* p = ws + off;
        off += (bytes + 255) & ~(size_t)255;
        return p;
    };
    bf16_t* xb    = (bf16_t*)carve((size_t)M * CDIM * 2);
    bf16_t* Wqb   = (bf16_t*)carve((size_t)CDIM * CDIM * 2);
    bf16_t* Wkb   = (bf16_t*)carve((size_t)CDIM * DDIM * 2);
    bf16_t* Wvb   = (bf16_t*)carve((size_t)CDIM * DDIM * 2);
    bf16_t* Wob   = (bf16_t*)carve((size_t)CDIM * CDIM * 2);
    bf16_t* qproj = (bf16_t*)carve((size_t)M * CDIM * 2);
    bf16_t* kproj = (bf16_t*)carve((size_t)M * DDIM * 2);
    bf16_t* vproj = (bf16_t*)carve((size_t)M * DDIM * 2);
    bf16_t* qrope = (bf16_t*)carve((size_t)M * CDIM * 2);
    bf16_t* krope = (bf16_t*)carve((size_t)M * DDIM * 2);
    bf16_t* attn  = qproj;  // reuse: qproj is dead after rope_q_k

    auto cvt = [&](const float* src, bf16_t* dst, int n) {
        cvt_f32_bf16_k<<<(n + 255) / 256, 256, 0, stream>>>(src, dst, n);
    };
    cvt(x,  xb,  M * CDIM);
    cvt(Wq, Wqb, CDIM * CDIM);
    cvt(Wk, Wkb, CDIM * DDIM);
    cvt(Wv, Wvb, CDIM * DDIM);
    cvt(Wo, Wob, CDIM * CDIM);

    dim3 blk(128);
    // projections
    gemm_bf16_k<false><<<dim3(CDIM / 64, M / 64), blk, 0, stream>>>(
        xb, Wqb, qproj, nullptr, M, CDIM, CDIM);
    gemm_bf16_k<false><<<dim3(DDIM / 64, M / 64), blk, 0, stream>>>(
        xb, Wkb, kproj, nullptr, M, DDIM, CDIM);
    gemm_bf16_k<false><<<dim3(DDIM / 64, M / 64), blk, 0, stream>>>(
        xb, Wvb, vproj, nullptr, M, DDIM, CDIM);
    // RoPE (+scale into q, head transpose into [B,H,T,D])
    rope_q_k<<<(BDIM * TDIM * N_HEADS * 32) / 256, 256, 0, stream>>>(qproj, qrope);
    rope_k_k<<<(BDIM * TDIM * 32) / 256, 256, 0, stream>>>(kproj, krope);
    // flash attention
    fa_mqa_k<<<dim3(TDIM / 64, N_HEADS, BDIM), blk, 0, stream>>>(
        qrope, krope, vproj, attn);
    // output projection -> f32
    gemm_bf16_k<true><<<dim3(CDIM / 64, M / 64), blk, 0, stream>>>(
        attn, Wob, nullptr, out, M, CDIM, CDIM);
}
